// I64TokenRoutedMLP_41420664603009
// MI455X (gfx1250) — compile-verified, matrix operand order checked by reading, not run on hardware
//
#include <hip/hip_runtime.h>
#include <hip/hip_bf16.h>

#define HIDDEN        2048
#define EXPERT_INTER  1024
#define NUM_EXPERTS   8
#define NUM_TOKENS    2048
#define VOCAB         32000

typedef __attribute__((ext_vector_type(16))) __bf16 v16bf;
typedef __attribute__((ext_vector_type(8)))  __bf16 v8bf;
typedef __attribute__((ext_vector_type(8)))  float  v8f;
typedef __attribute__((ext_vector_type(4)))  int    v4i;

union F16 {          // one WMMA 16-bf16 operand = two 16B LDS loads
  v16bf v;
  uint4 q[2];
};

// Async global->LDS copy path (gfx1250, ASYNCcnt) if the toolchain declares it.
// ROCm clang-22 signature (from probe diagnostics): params are v4i* pointers.
#if defined(__has_builtin)
#if __has_builtin(__builtin_amdgcn_global_load_async_to_lds_b128) && \
    __has_builtin(__builtin_amdgcn_s_wait_asynccnt)
#define USE_ASYNC_LDS 1
#endif
#endif

#define VP(p) ((v4i*)(p))   // C-style cast: drops const, reinterprets to v4i*

// ---------------------------------------------------------------------------
// Kernel 1: routing. expert = clip(id) % 8. Build segment offsets + permutation.
// ---------------------------------------------------------------------------
__global__ __launch_bounds__(256) void route_kernel(
    const int* __restrict__ token_ids, int* __restrict__ perm,
    int* __restrict__ seg_off) {
  __shared__ int cnt[NUM_EXPERTS];
  __shared__ int cur[NUM_EXPERTS];
  const int tid = threadIdx.x;
  if (tid < NUM_EXPERTS) cnt[tid] = 0;
  __syncthreads();
  for (int t = tid; t < NUM_TOKENS; t += blockDim.x) {
    int id = token_ids[t];
    id = id < 0 ? 0 : (id > VOCAB - 1 ? VOCAB - 1 : id);
    atomicAdd(&cnt[id % NUM_EXPERTS], 1);
  }
  __syncthreads();
  if (tid == 0) {
    int acc = 0;
    for (int e = 0; e < NUM_EXPERTS; ++e) {
      cur[e] = acc;
      seg_off[e] = acc;
      acc += cnt[e];
    }
    seg_off[NUM_EXPERTS] = acc;
  }
  __syncthreads();
  for (int t = tid; t < NUM_TOKENS; t += blockDim.x) {
    int id = token_ids[t];
    id = id < 0 ? 0 : (id > VOCAB - 1 ? VOCAB - 1 : id);
    const int e = id % NUM_EXPERTS;
    const int pos = atomicAdd(&cur[e], 1);
    perm[pos] = t;
  }
}

// ---------------------------------------------------------------------------
// Kernel 2: gather x rows into expert-sorted bf16 buffer xg[2048][2048].
// ---------------------------------------------------------------------------
__global__ __launch_bounds__(256) void gather_kernel(
    const float* __restrict__ x, const int* __restrict__ perm,
    __bf16* __restrict__ xg) {
  const int r = blockIdx.x;
  const int src = perm[r];
  const float4* __restrict__ xin =
      reinterpret_cast<const float4*>(x + (size_t)src * HIDDEN);
  __bf16* __restrict__ orow = xg + (size_t)r * HIDDEN;
  const int t = threadIdx.x;            // 256 threads * 8 elems = 2048
  const float4 v0 = xin[t * 2 + 0];
  const float4 v1 = xin[t * 2 + 1];
  v8bf h;
  h[0] = (__bf16)v0.x; h[1] = (__bf16)v0.y; h[2] = (__bf16)v0.z; h[3] = (__bf16)v0.w;
  h[4] = (__bf16)v1.x; h[5] = (__bf16)v1.y; h[6] = (__bf16)v1.z; h[7] = (__bf16)v1.w;
  *reinterpret_cast<v8bf*>(orow + t * 8) = h;
}

// ---------------------------------------------------------------------------
// WMMA fragment layout (wave32, 16x16x32 bf16):
//   lane = hl*16 + lr;  A: lr = M row, B: lr = N col, hl selects K-half.
//   A lane fragment = As[row][ki + hl*8 .. +7]  and  As[row][ki+16+hl*8 .. +7]
//   (two contiguous ascending-K runs -> two ds_load_b128).
//   B tile stored TRANSPOSED in LDS (Bt[n][k]) so B fragments are identical.
//   C/D: lane lr = N col; VGPR i -> M = i + 8*hl.
// Block tile: M=64, N=64, K-step 64. 256 threads = 8 waves (4 M x 2 N),
// each wave owns two 16x16 output tiles (nw*32 + {0,16}).
// ---------------------------------------------------------------------------

// ---------------------------------------------------------------------------
// Kernel 3: GEMM1 + fused SiLU*up.
// ---------------------------------------------------------------------------
__global__ __launch_bounds__(256) void moe_gemm1(
    const __bf16* __restrict__ xg, const float* __restrict__ gate_up,
    const int* __restrict__ seg_off, __bf16* __restrict__ inter) {
  const int e     = blockIdx.z;
  const int mtile = blockIdx.y;
  const int ntile = blockIdx.x;
  const int seg_start = seg_off[e];
  const int seg_rows  = seg_off[e + 1] - seg_start;
  if (mtile * 64 >= seg_rows) return;

  __shared__ __align__(16) __bf16 As[64][72];   // [m][k], 144B rows
  __shared__ __align__(16) __bf16 Bg[64][72];   // transposed: [n][k]
  __shared__ __align__(16) __bf16 Bu[64][72];   // transposed: [n][k]

  const int tid  = threadIdx.x;
  const int wave = tid >> 5;
  const int lane = tid & 31;
  const int hl   = lane >> 4;
  const int lr   = lane & 15;
  const int mw   = wave & 3;
  const int nw   = wave >> 2;

  const float* __restrict__ W =
      gate_up + (size_t)e * HIDDEN * (2 * EXPERT_INTER);
  const int n0  = ntile * 64;
  const int ldw = 2 * EXPERT_INTER;

  // staging thread mapping
  const int ar  = tid >> 2;           // A: row 0..63
  const int acs = (tid & 3) * 16;     // A: k offset 0,16,32,48
  const int bn  = tid & 63;           // B: n 0..63
  const int bks = (tid >> 6) * 16;    // B: k offset 0,16,32,48

  v8f cg[2], cu[2];
#pragma unroll
  for (int t = 0; t < 2; ++t)
#pragma unroll
    for (int i = 0; i < 8; ++i) { cg[t][i] = 0.0f; cu[t][i] = 0.0f; }

  for (int k0 = 0; k0 < HIDDEN; k0 += 64) {
    // ---- stage A (64x64 bf16): async global->LDS (no VGPR roundtrip) ----
    {
      const int grow = mtile * 64 + ar;
      if (grow < seg_rows) {
        const __bf16* p = xg + (size_t)(seg_start + grow) * HIDDEN + k0 + acs;
#if USE_ASYNC_LDS
        __builtin_amdgcn_global_load_async_to_lds_b128(
            VP(p), VP(&As[ar][acs]), 0, 0);
        __builtin_amdgcn_global_load_async_to_lds_b128(
            VP(p + 8), VP(&As[ar][acs + 8]), 0, 0);
#else
        const uint4 v0 = reinterpret_cast<const uint4*>(p)[0];
        const uint4 v1 = reinterpret_cast<const uint4*>(p)[1];
        *reinterpret_cast<uint4*>(&As[ar][acs])     = v0;
        *reinterpret_cast<uint4*>(&As[ar][acs + 8]) = v1;
#endif
      } else {
        const uint4 z = make_uint4(0u, 0u, 0u, 0u);
        *reinterpret_cast<uint4*>(&As[ar][acs])     = z;
        *reinterpret_cast<uint4*>(&As[ar][acs + 8]) = z;
      }
    }
    // ---- stage B gate & up transposed: column loads (coalesced across lanes),
    //      convert fp32->bf16 (v_cvt_pk_bf16_f32), vector LDS stores ----
    {
      const float* __restrict__ pg = W + (size_t)(k0 + bks) * ldw + n0 + bn;
      const float* __restrict__ pu = pg + EXPERT_INTER;
      if (k0 + 64 < HIDDEN) {
        __builtin_prefetch(pg + (size_t)64 * ldw, 0, 1);
        __builtin_prefetch(pu + (size_t)64 * ldw, 0, 1);
      }
      v8bf hg0, hg1, hu0, hu1;
#pragma unroll
      for (int j = 0; j < 8; ++j) {
        hg0[j] = (__bf16)pg[(size_t)j * ldw];
        hg1[j] = (__bf16)pg[(size_t)(j + 8) * ldw];
        hu0[j] = (__bf16)pu[(size_t)j * ldw];
        hu1[j] = (__bf16)pu[(size_t)(j + 8) * ldw];
      }
      *reinterpret_cast<v8bf*>(&Bg[bn][bks])     = hg0;
      *reinterpret_cast<v8bf*>(&Bg[bn][bks + 8]) = hg1;
      *reinterpret_cast<v8bf*>(&Bu[bn][bks])     = hu0;
      *reinterpret_cast<v8bf*>(&Bu[bn][bks + 8]) = hu1;
    }
#if USE_ASYNC_LDS
    __builtin_amdgcn_s_wait_asynccnt(0);
#endif
    __syncthreads();

    const int arow = mw * 16 + lr;
#pragma unroll
    for (int ki = 0; ki < 64; ki += 32) {
      F16 a;
      a.q[0] = *reinterpret_cast<const uint4*>(&As[arow][ki + hl * 8]);
      a.q[1] = *reinterpret_cast<const uint4*>(&As[arow][ki + 16 + hl * 8]);
#pragma unroll
      for (int t = 0; t < 2; ++t) {
        const int bcol = nw * 32 + t * 16 + lr;
        F16 bg, bu;
        bg.q[0] = *reinterpret_cast<const uint4*>(&Bg[bcol][ki + hl * 8]);
        bg.q[1] = *reinterpret_cast<const uint4*>(&Bg[bcol][ki + 16 + hl * 8]);
        bu.q[0] = *reinterpret_cast<const uint4*>(&Bu[bcol][ki + hl * 8]);
        bu.q[1] = *reinterpret_cast<const uint4*>(&Bu[bcol][ki + 16 + hl * 8]);
        cg[t] = __builtin_amdgcn_wmma_f32_16x16x32_bf16(
            false, a.v, false, bg.v, (short)0, cg[t], false, false);
        cu[t] = __builtin_amdgcn_wmma_f32_16x16x32_bf16(
            false, a.v, false, bu.v, (short)0, cu[t], false, false);
      }
    }
    __syncthreads();
  }

  // ---- epilogue: inter = silu(gate) * up, bf16 store ----
#pragma unroll
  for (int t = 0; t < 2; ++t) {
    const int col = n0 + nw * 32 + t * 16 + lr;
#pragma unroll
    for (int i = 0; i < 8; ++i) {
      const int rl   = mw * 16 + hl * 8 + i;
      const int grow = mtile * 64 + rl;
      if (grow < seg_rows) {
        const float g = cg[t][i];
        const float s = g / (1.0f + __expf(-g));
        inter[(size_t)(seg_start + grow) * EXPERT_INTER + col] =
            (__bf16)(s * cu[t][i]);
      }
    }
  }
}

// ---------------------------------------------------------------------------
// Kernel 4: GEMM2, out[perm[row]][n] = inter_row . down[:,n]  (fp32 out)
// ---------------------------------------------------------------------------
__global__ __launch_bounds__(256) void moe_gemm2(
    const __bf16* __restrict__ inter, const float* __restrict__ down,
    const int* __restrict__ seg_off, const int* __restrict__ perm,
    float* __restrict__ out) {
  const int e     = blockIdx.z;
  const int mtile = blockIdx.y;
  const int ntile = blockIdx.x;
  const int seg_start = seg_off[e];
  const int seg_rows  = seg_off[e + 1] - seg_start;
  if (mtile * 64 >= seg_rows) return;

  __shared__ __align__(16) __bf16 As[64][72];
  __shared__ __align__(16) __bf16 Bt[64][72];   // transposed: [n][k]

  const int tid  = threadIdx.x;
  const int wave = tid >> 5;
  const int lane = tid & 31;
  const int hl   = lane >> 4;
  const int lr   = lane & 15;
  const int mw   = wave & 3;
  const int nw   = wave >> 2;

  const float* __restrict__ W = down + (size_t)e * EXPERT_INTER * HIDDEN;
  const int n0 = ntile * 64;

  const int ar  = tid >> 2;
  const int acs = (tid & 3) * 16;
  const int bn  = tid & 63;
  const int bks = (tid >> 6) * 16;

  v8f c[2];
#pragma unroll
  for (int t = 0; t < 2; ++t)
#pragma unroll
    for (int i = 0; i < 8; ++i) c[t][i] = 0.0f;

  for (int k0 = 0; k0 < EXPERT_INTER; k0 += 64) {
    {
      const int grow = mtile * 64 + ar;
      if (grow < seg_rows) {
        const __bf16* p =
            inter + (size_t)(seg_start + grow) * EXPERT_INTER + k0 + acs;
#if USE_ASYNC_LDS
        __builtin_amdgcn_global_load_async_to_lds_b128(
            VP(p), VP(&As[ar][acs]), 0, 0);
        __builtin_amdgcn_global_load_async_to_lds_b128(
            VP(p + 8), VP(&As[ar][acs + 8]), 0, 0);
#else
        const uint4 v0 = reinterpret_cast<const uint4*>(p)[0];
        const uint4 v1 = reinterpret_cast<const uint4*>(p)[1];
        *reinterpret_cast<uint4*>(&As[ar][acs])     = v0;
        *reinterpret_cast<uint4*>(&As[ar][acs + 8]) = v1;
#endif
      } else {
        const uint4 z = make_uint4(0u, 0u, 0u, 0u);
        *reinterpret_cast<uint4*>(&As[ar][acs])     = z;
        *reinterpret_cast<uint4*>(&As[ar][acs + 8]) = z;
      }
    }
    {
      const float* __restrict__ pw = W + (size_t)(k0 + bks) * HIDDEN + n0 + bn;
      if (k0 + 64 < EXPERT_INTER)
        __builtin_prefetch(pw + (size_t)64 * HIDDEN, 0, 1);
      v8bf h0, h1;
#pragma unroll
      for (int j = 0; j < 8; ++j) {
        h0[j] = (__bf16)pw[(size_t)j * HIDDEN];
        h1[j] = (__bf16)pw[(size_t)(j + 8) * HIDDEN];
      }
      *reinterpret_cast<v8bf*>(&Bt[bn][bks])     = h0;
      *reinterpret_cast<v8bf*>(&Bt[bn][bks + 8]) = h1;
    }
#if USE_ASYNC_LDS
    __builtin_amdgcn_s_wait_asynccnt(0);
#endif
    __syncthreads();

    const int arow = mw * 16 + lr;
#pragma unroll
    for (int ki = 0; ki < 64; ki += 32) {
      F16 a;
      a.q[0] = *reinterpret_cast<const uint4*>(&As[arow][ki + hl * 8]);
      a.q[1] = *reinterpret_cast<const uint4*>(&As[arow][ki + 16 + hl * 8]);
#pragma unroll
      for (int t = 0; t < 2; ++t) {
        const int bcol = nw * 32 + t * 16 + lr;
        F16 b;
        b.q[0] = *reinterpret_cast<const uint4*>(&Bt[bcol][ki + hl * 8]);
        b.q[1] = *reinterpret_cast<const uint4*>(&Bt[bcol][ki + 16 + hl * 8]);
        c[t] = __builtin_amdgcn_wmma_f32_16x16x32_bf16(
            false, a.v, false, b.v, (short)0, c[t], false, false);
      }
    }
    __syncthreads();
  }

  // ---- epilogue: scatter fp32 rows to original token slots ----
#pragma unroll
  for (int t = 0; t < 2; ++t) {
    const int col = n0 + nw * 32 + t * 16 + lr;
#pragma unroll
    for (int i = 0; i < 8; ++i) {
      const int rl   = mw * 16 + hl * 8 + i;
      const int grow = mtile * 64 + rl;
      if (grow < seg_rows) {
        const int tok = perm[seg_start + grow];
        out[(size_t)tok * HIDDEN + col] = c[t][i];
      }
    }
  }
}

// ---------------------------------------------------------------------------
// Host launcher
// ---------------------------------------------------------------------------
extern "C" void kernel_launch(void* const* d_in, const int* in_sizes, int n_in,
                              void* d_out, int out_size, void* d_ws,
                              size_t ws_size, hipStream_t stream) {
  const float* x         = (const float*)d_in[0];
  const int*   token_ids = (const int*)d_in[1];
  const float* gate_up   = (const float*)d_in[2];
  const float* down      = (const float*)d_in[3];
  float*       out       = (float*)d_out;

  char* ws = (char*)d_ws;
  int*     perm  = (int*)ws;                               // 2048 * 4 B
  int*     seg   = (int*)(ws + NUM_TOKENS * 4);            // 9 * 4 B (pad 64)
  __bf16*  xg    = (__bf16*)(ws + NUM_TOKENS * 4 + 64);    // 8 MB
  __bf16*  inter = (__bf16*)((char*)xg + (size_t)NUM_TOKENS * HIDDEN * 2);  // 4 MB

  route_kernel<<<1, 256, 0, stream>>>(token_ids, perm, seg);
  gather_kernel<<<NUM_TOKENS, 256, 0, stream>>>(x, perm, xg);

  dim3 g1(EXPERT_INTER / 64, NUM_TOKENS / 64, NUM_EXPERTS);
  moe_gemm1<<<g1, 256, 0, stream>>>(xg, gate_up, seg, inter);

  dim3 g2(HIDDEN / 64, NUM_TOKENS / 64, NUM_EXPERTS);
  moe_gemm2<<<g2, 256, 0, stream>>>(inter, down, seg, perm, out);
}
